// InfiniMultiHeadAttention_14946486190746
// MI455X (gfx1250) — matrix operand order repaired and checked by reading, not compiled, not run on hardware
//
#include <hip/hip_runtime.h>
#include <hip/hip_bf16.h>

typedef __bf16 bf16_t;
typedef __attribute__((ext_vector_type(16))) bf16_t        bf16x16;
typedef __attribute__((ext_vector_type(8)))  float         floatx8;
typedef __attribute__((ext_vector_type(8)))  unsigned short ushort8;
typedef __attribute__((ext_vector_type(16))) unsigned short ushort16;

// ---------------- bf16 helpers (RNE-ish) ----------------
__device__ __forceinline__ unsigned short f32_to_bf16(float f) {
  unsigned u = __builtin_bit_cast(unsigned, f);
  unsigned r = u + 0x7FFFu + ((u >> 16) & 1u);
  return (unsigned short)(r >> 16);
}
__device__ __forceinline__ float bf16_to_f32(unsigned short h) {
  unsigned u = ((unsigned)h) << 16;
  return __builtin_bit_cast(float, u);
}
__device__ __forceinline__ floatx8 zero8() {
  floatx8 z = {0.f, 0.f, 0.f, 0.f, 0.f, 0.f, 0.f, 0.f};
  return z;
}

// ---------------- WMMA fragment loads (wave32) ----------------
// A: 16x32 bf16, row-major source [*, ld].
// lane<16: row=lane,   K = {0..7, 16..23}
// lane>=16: row=lane-16, K = {8..15, 24..31}
__device__ __forceinline__ bf16x16 load_frag_a(const unsigned short* base, int ld) {
  int lane = threadIdx.x & 31;
  int row  = lane & 15;
  int kb   = (lane >> 4) << 3;          // 0 or 8
  const unsigned short* p = base + (size_t)row * ld + kb;
  ushort8 lo = *(const ushort8*)(p);      // K = kb .. kb+7
  ushort8 hi = *(const ushort8*)(p + 16); // K = kb+16 .. kb+23
  ushort16 u = __builtin_shufflevector(lo, hi, 0,1,2,3,4,5,6,7,8,9,10,11,12,13,14,15);
  return __builtin_bit_cast(bf16x16, u);
}

// B: 32x16 bf16, source supplied TRANSPOSED (Bt is [N x K] row-major).
// lane<16: col=lane, K = {0..7, 16..23} ; lane>=16: col=lane-16, K = {8..15, 24..31}
__device__ __forceinline__ bf16x16 load_frag_bT(const unsigned short* baseT, int ld) {
  int lane = threadIdx.x & 31;
  int col  = lane & 15;
  int kb   = (lane >> 4) << 3;          // 0 or 8
  const unsigned short* p = baseT + (size_t)col * ld + kb;
  ushort8 lo = *(const ushort8*)(p);
  ushort8 hi = *(const ushort8*)(p + 16);
  ushort16 u = __builtin_shufflevector(lo, hi, 0,1,2,3,4,5,6,7,8,9,10,11,12,13,14,15);
  return __builtin_bit_cast(bf16x16, u);
}

__device__ __forceinline__ floatx8 wmma_bf16(bf16x16 a, bf16x16 b, floatx8 c) {
  return __builtin_amdgcn_wmma_f32_16x16x32_bf16(false, a, false, b, (short)0, c,
                                                 false, false);
}

// ---------------- conversion / transpose kernels ----------------
__global__ void cvt_f32_bf16_kernel(const float* __restrict__ src,
                                    unsigned short* __restrict__ dst, long long n) {
  long long i = (long long)blockIdx.x * blockDim.x + threadIdx.x;
  long long stride = (long long)gridDim.x * blockDim.x;
  for (; i < n; i += stride) dst[i] = f32_to_bf16(src[i]);
}

// W is [K x N] row-major f32 -> WT is [N x K] row-major bf16
__global__ void transpose_to_bf16_kernel(const float* __restrict__ w,
                                         unsigned short* __restrict__ wT,
                                         int K, int N) {
  int n = blockIdx.x * 16 + threadIdx.x;
  int k = blockIdx.y * 16 + threadIdx.y;
  if (n < N && k < K) wT[(size_t)n * K + k] = f32_to_bf16(w[(size_t)k * N + n]);
}

// ---------------- fused QKV projection GEMM ----------------
// M=32768, N=1024, K=1024. Block: 128 threads = 4 waves; wave -> 16x64 strip.
__global__ __launch_bounds__(128)
void proj_qkv_kernel(const unsigned short* __restrict__ xb,
                     const unsigned short* __restrict__ wqT,
                     const unsigned short* __restrict__ wkT,
                     const unsigned short* __restrict__ wvT,
                     unsigned short* __restrict__ qb,
                     unsigned short* __restrict__ kbuf,
                     unsigned short* __restrict__ vbuf) {
  const int wave = threadIdx.x >> 5;
  const int m0 = blockIdx.x * 64 + wave * 16;
  const int n0 = blockIdx.y * 64;
  floatx8 aq[4], ak[4], av[4];
  for (int t = 0; t < 4; t++) { aq[t] = zero8(); ak[t] = zero8(); av[t] = zero8(); }
  for (int kk = 0; kk < 1024; kk += 32) {
    bf16x16 a = load_frag_a(xb + (size_t)m0 * 1024 + kk, 1024);
#pragma unroll
    for (int t = 0; t < 4; t++) {
      const size_t off = (size_t)(n0 + t * 16) * 1024 + kk;
      aq[t] = wmma_bf16(a, load_frag_bT(wqT + off, 1024), aq[t]);
      ak[t] = wmma_bf16(a, load_frag_bT(wkT + off, 1024), ak[t]);
      av[t] = wmma_bf16(a, load_frag_bT(wvT + off, 1024), av[t]);
    }
  }
  const int lane = threadIdx.x & 31;
  const int cn = lane & 15;
  const int rm = (lane >> 4) << 3;
  for (int t = 0; t < 4; t++)
#pragma unroll
    for (int r = 0; r < 8; r++) {
      size_t o = (size_t)(m0 + rm + r) * 1024 + (size_t)(n0 + t * 16 + cn);
      qb[o]   = f32_to_bf16(aq[t][r]);
      kbuf[o] = f32_to_bf16(ak[t][r]);
      vbuf[o] = f32_to_bf16(av[t][r]);
    }
}

// ---------------- output GEMM: att(bf16) @ WoT -> f32 out ----------------
__global__ __launch_bounds__(128)
void out_gemm_kernel(const unsigned short* __restrict__ ab,
                     const unsigned short* __restrict__ woT,
                     float* __restrict__ out) {
  const int wave = threadIdx.x >> 5;
  const int m0 = blockIdx.x * 64 + wave * 16;
  const int n0 = blockIdx.y * 64;
  floatx8 acc[4];
  for (int t = 0; t < 4; t++) acc[t] = zero8();
  for (int kk = 0; kk < 1024; kk += 32) {
    bf16x16 a = load_frag_a(ab + (size_t)m0 * 1024 + kk, 1024);
#pragma unroll
    for (int t = 0; t < 4; t++) {
      const size_t off = (size_t)(n0 + t * 16) * 1024 + kk;
      acc[t] = wmma_bf16(a, load_frag_bT(woT + off, 1024), acc[t]);
    }
  }
  const int lane = threadIdx.x & 31;
  const int cn = lane & 15;
  const int rm = (lane >> 4) << 3;
  for (int t = 0; t < 4; t++)
#pragma unroll
    for (int r = 0; r < 8; r++)
      out[(size_t)(m0 + rm + r) * 1024 + (size_t)(n0 + t * 16 + cn)] = acc[t][r];
}

// ---------------- attention + linear-memory kernel ----------------
// One workgroup (256 thr = 8 waves) per (b,h), sequential over 16 segments.
__global__ __launch_bounds__(256)
void attn_kernel(const unsigned short* __restrict__ qb,
                 const unsigned short* __restrict__ kb,
                 const unsigned short* __restrict__ vb,
                 const float* __restrict__ betas,
                 const float* __restrict__ memory0,
                 const float* __restrict__ z0,
                 unsigned short* __restrict__ attb,
                 float* __restrict__ sc_ws,           // per bh: 512*512 f32
                 unsigned short* __restrict__ p_ws,   // per bh: 512*512 bf16
                 unsigned short* __restrict__ vt_ws,  // per bh: 64*512 bf16
                 float* __restrict__ ad_ws)           // per bh: 512*64 f32
{
  const int bh = blockIdx.x;
  const int b = bh >> 4, h = bh & 15;
  const int tid = threadIdx.x;
  const int wave = tid >> 5;

  __shared__ float M_lds[64 * 64];
  __shared__ float Z_lds[64];
  __shared__ float gate_lds[64];

  for (int i = tid; i < 4096; i += 256) M_lds[i] = memory0[h * 4096 + i];
  if (tid < 64) {
    Z_lds[tid] = z0[h * 64 + tid];
    gate_lds[tid] = 1.0f / (1.0f + __expf(-betas[h * 64 + tid]));
  }
  __syncthreads();

  float*          sc   = sc_ws + (size_t)bh * 512 * 512;
  unsigned short* pmat = p_ws  + (size_t)bh * 512 * 512;
  unsigned short* vt   = vt_ws + (size_t)bh * 64 * 512;
  float*          ad   = ad_ws + (size_t)bh * 512 * 64;
  const float scale = 0.125f;  // 1/sqrt(64)

  for (int seg = 0; seg < 16; seg++) {
    const size_t blk = ((size_t)(b * 8192 + seg * 512)) * 1024 + (size_t)h * 512 * 64;
    const unsigned short* qh = qb + blk;
    const unsigned short* kh = kb + blk;
    const unsigned short* vh = vb + blk;

    // phase 0: V^T (64 x 512) for B-fragment-friendly loads
    for (int i = tid; i < 512 * 64; i += 256) {
      int l = i >> 6, d = i & 63;
      vt[(size_t)d * 512 + l] = vh[(size_t)l * 64 + d];
    }
    __syncthreads();

    // phase A: causal score tiles S = Q K^T * scale (lower triangle only)
    {
      int t = 0;
      for (int i = 0; i < 32; i++)
        for (int j = 0; j <= i; j++, t++) {
          if ((t & 7) != wave) continue;
          floatx8 acc = zero8();
#pragma unroll
          for (int kk = 0; kk < 64; kk += 32) {
            bf16x16 a  = load_frag_a(qh + (size_t)(i * 16) * 64 + kk, 64);
            bf16x16 bm = load_frag_bT(kh + (size_t)(j * 16) * 64 + kk, 64);
            acc = wmma_bf16(a, bm, acc);
          }
          const int lane = tid & 31;
          const int cn = j * 16 + (lane & 15);
          const int rm = i * 16 + ((lane >> 4) << 3);
#pragma unroll
          for (int r = 0; r < 8; r++) {
            float s = acc[r] * scale;
            if (cn > rm + r) s = -3.0e38f;  // strictly-upper masked
            sc[(size_t)(rm + r) * 512 + cn] = s;
          }
        }
    }
    __syncthreads();

    // phase B: row softmax (valid cols 0..row), write zero-padded bf16 P
    for (int rr = 0; rr < 2; rr++) {
      int row = tid + rr * 256;
      float mx = -3.0e38f;
      for (int c = 0; c <= row; c++) mx = fmaxf(mx, sc[(size_t)row * 512 + c]);
      float sum = 0.f;
      for (int c = 0; c <= row; c++) sum += __expf(sc[(size_t)row * 512 + c] - mx);
      float inv = 1.0f / sum;
      for (int c = 0; c < 512; c++) {
        float p = (c <= row) ? __expf(sc[(size_t)row * 512 + c] - mx) * inv : 0.0f;
        pmat[(size_t)row * 512 + c] = f32_to_bf16(p);
      }
    }
    __syncthreads();

    // phase C: att_dot = P @ V  (K bounded by causal extent, zeros pad rest)
    for (int tt = wave; tt < 128; tt += 8) {
      int i = tt >> 2, jn = tt & 3;
      floatx8 acc = zero8();
      int kend = ((i + 2) >> 1) << 5;  // ceil32((i+1)*16)
      for (int kk = 0; kk < kend; kk += 32) {
        bf16x16 a  = load_frag_a(pmat + (size_t)(i * 16) * 512 + kk, 512);
        bf16x16 bm = load_frag_bT(vt + (size_t)(jn * 16) * 512 + kk, 512);
        acc = wmma_bf16(a, bm, acc);
      }
      const int lane = tid & 31;
      const int cn = jn * 16 + (lane & 15);
      const int rm = i * 16 + ((lane >> 4) << 3);
#pragma unroll
      for (int r = 0; r < 8; r++) ad[(size_t)(rm + r) * 64 + cn] = acc[r];
    }
    __syncthreads();

    // phase D: memory retrieval with OLD M/Z, gated combine, store att (bf16)
    for (int rr = 0; rr < 2; rr++) {
      int l = tid + rr * 256;
      float sq[64];
      float denom = 0.f;
      for (int d = 0; d < 64; d++) {
        float qv = bf16_to_f32(qh[(size_t)l * 64 + d]);
        float e = (qv > 0.f) ? (qv + 1.0f) : __expf(qv);  // elu(x)+1
        sq[d] = e;
        denom += e * Z_lds[d];
      }
      float invden = 1.0f / denom;
      for (int d = 0; d < 64; d++) {
        float num = 0.f;
        for (int j = 0; j < 64; j++) num += sq[j] * M_lds[j * 64 + d];
        float g = gate_lds[d];
        float a = g * (num * invden) + (1.f - g) * ad[(size_t)l * 64 + d];
        attb[blk + (size_t)l * 64 + d] = f32_to_bf16(a);
      }
    }
    __syncthreads();

    // phase E: M += sk^T @ v ; Z += sum_l sk
    {
      int d = tid >> 2;
      int e0 = (tid & 3) << 4;
      float accm[16];
#pragma unroll
      for (int e = 0; e < 16; e++) accm[e] = 0.f;
      float accz = 0.f;
      for (int l = 0; l < 512; l++) {
        float kv = bf16_to_f32(kh[(size_t)l * 64 + d]);
        float sk = (kv > 0.f) ? (kv + 1.0f) : __expf(kv);
        if ((tid & 3) == 0) accz += sk;
#pragma unroll
        for (int e = 0; e < 16; e++)
          accm[e] += sk * bf16_to_f32(vh[(size_t)l * 64 + e0 + e]);
      }
      for (int e = 0; e < 16; e++) M_lds[d * 64 + e0 + e] += accm[e];
      if ((tid & 3) == 0) Z_lds[d] += accz;
    }
    __syncthreads();
  }
}

// ---------------- host-side orchestration ----------------
extern "C" void kernel_launch(void* const* d_in, const int* in_sizes, int n_in,
                              void* d_out, int out_size, void* d_ws, size_t ws_size,
                              hipStream_t stream) {
  (void)in_sizes; (void)n_in; (void)out_size; (void)ws_size;
  const float* x       = (const float*)d_in[0];
  const float* Wq      = (const float*)d_in[1];
  const float* Wk      = (const float*)d_in[2];
  const float* Wv      = (const float*)d_in[3];
  const float* Wo      = (const float*)d_in[4];
  const float* betas   = (const float*)d_in[5];
  const float* memory0 = (const float*)d_in[6];
  const float* z0      = (const float*)d_in[7];
  float* out = (float*)d_out;

  const long long NTOK = 32768;      // B*S
  const long long DMOD = 1024;       // D = H*DK = H*DV

  char* ws = (char*)d_ws;
  size_t off = 0;
  auto alloc = [&](size_t bytes) -> void* {
    void* p = (void*)(ws + off);
    off += (bytes + 255) & ~(size_t)255;
    return p;
  };
  unsigned short* xb   = (unsigned short*)alloc((size_t)NTOK * DMOD * 2);
  unsigned short* wqT  = (unsigned short*)alloc((size_t)DMOD * DMOD * 2);
  unsigned short* wkT  = (unsigned short*)alloc((size_t)DMOD * DMOD * 2);
  unsigned short* wvT  = (unsigned short*)alloc((size_t)DMOD * DMOD * 2);
  unsigned short* woT  = (unsigned short*)alloc((size_t)DMOD * DMOD * 2);
  unsigned short* qb   = (unsigned short*)alloc((size_t)NTOK * DMOD * 2);
  unsigned short* kbuf = (unsigned short*)alloc((size_t)NTOK * DMOD * 2);
  unsigned short* vbuf = (unsigned short*)alloc((size_t)NTOK * DMOD * 2);
  unsigned short* attb = (unsigned short*)alloc((size_t)NTOK * DMOD * 2);
  float*          sc   = (float*)alloc((size_t)64 * 512 * 512 * 4);
  unsigned short* pmat = (unsigned short*)alloc((size_t)64 * 512 * 512 * 2);
  unsigned short* vt   = (unsigned short*)alloc((size_t)64 * 64 * 512 * 2);
  float*          ad   = (float*)alloc((size_t)64 * 512 * 64 * 4);

  // 1) convert x -> bf16 ; weights -> bf16 transposed
  cvt_f32_bf16_kernel<<<2048, 256, 0, stream>>>(x, xb, NTOK * DMOD);
  dim3 tg(64, 64), tb(16, 16);
  transpose_to_bf16_kernel<<<tg, tb, 0, stream>>>(Wq, wqT, 1024, 1024);
  transpose_to_bf16_kernel<<<tg, tb, 0, stream>>>(Wk, wkT, 1024, 1024);
  transpose_to_bf16_kernel<<<tg, tb, 0, stream>>>(Wv, wvT, 1024, 1024);
  transpose_to_bf16_kernel<<<tg, tb, 0, stream>>>(Wo, woT, 1024, 1024);

  // 2) fused QKV projection (WMMA bf16)
  proj_qkv_kernel<<<dim3(512, 16), 128, 0, stream>>>(xb, wqT, wkT, wvT, qb, kbuf, vbuf);

  // 3) per-(b,h) attention with sequential segment scan (WMMA bf16 + LDS state)
  attn_kernel<<<64, 256, 0, stream>>>(qb, kbuf, vbuf, betas, memory0, z0,
                                      attb, sc, pmat, vt, ad);

  // 4) output projection (WMMA bf16 -> f32 out)
  out_gemm_kernel<<<dim3(512, 16), 128, 0, stream>>>(attb, woT, out);
}
